// MetaHopfieldNoGate_31825707664064
// MI455X (gfx1250) — matrix operand order repaired and checked by reading, not compiled, not run on hardware
//
#include <hip/hip_runtime.h>
#include <hip/hip_bf16.h>
#include <math.h>

// ---------------------------------------------------------------------------
// MetaHopfieldNoGate on MI455X (gfx1250).
// flash-attention x2 + WMMA bf16 GEMMs, wave32, 8 waves / block.
// All B operands are stored column-major (transposed) so every WMMA fragment
// is 1-2 contiguous 16/32B vector loads. Uniform LDS strip staging goes
// through the Tensor Data Mover (6-arg clang-23 builtin form).
// ---------------------------------------------------------------------------

typedef __bf16 bf16_t;
typedef __attribute__((ext_vector_type(16))) __bf16        v16bf;
typedef __attribute__((ext_vector_type(8)))  float         v8f;
typedef __attribute__((ext_vector_type(8)))  unsigned int  v8u;
typedef __attribute__((ext_vector_type(4)))  unsigned int  v4u;
typedef __attribute__((ext_vector_type(8)))  int           v8i;
typedef __attribute__((ext_vector_type(4)))  int           v4i;

#define NROWS   4096
#define MCTX    32768
#define DMODEL  768
#define DINNER  256
#define WPB     8        // waves per block
#define DVSLICE 96       // 768 / 8 waves
#define NTILES  6        // 96 / 16

#if defined(__has_builtin)
# if __has_builtin(__builtin_amdgcn_tensor_load_to_lds)
#  define USE_TDM 1
# endif
#endif
#ifndef USE_TDM
# define USE_TDM 0
#endif

__device__ __forceinline__ float gelu_exact(float x) {
  return 0.5f * x * (1.0f + erff(x * 0.70710678118654752f));
}
__device__ __forceinline__ float softplus_f(float x) {
  return (x > 20.0f) ? x : log1pf(__expf(x));
}
__device__ __forceinline__ v8f zero8() {
  v8f z = {0.f, 0.f, 0.f, 0.f, 0.f, 0.f, 0.f, 0.f};
  return z;
}

// 16 contiguous bf16 -> fragment half/whole via one 32B vector load.
__device__ __forceinline__ v16bf load_contig16(const bf16_t* p) {
  v8u u = *(const v8u*)p;
  return __builtin_bit_cast(v16bf, u);
}

// A fragment (16x32 bf16) from row-major src (stride lda), origin (row0,k0).
// lane row = lane&15; lanes 0-15 hold K 0..15, lanes 16-31 hold K 16..31
// (matches load_bt_frag). Two aligned 16B loads per lane.
__device__ __forceinline__ v16bf load_a_frag(const bf16_t* a, int lda,
                                             int row0, int k0, int lane) {
  const int row = row0 + (lane & 15);
  const bf16_t* p = a + (size_t)row * lda + k0 + (lane >> 4) * 16;
  v4u lo = *(const v4u*)p;
  v4u hi = *(const v4u*)(p + 8);
  v8u u;
  u[0] = lo[0]; u[1] = lo[1]; u[2] = lo[2]; u[3] = lo[3];
  u[4] = hi[0]; u[5] = hi[1]; u[6] = hi[2]; u[7] = hi[3];
  return __builtin_bit_cast(v16bf, u);
}

// B fragment (32x16) where logical B(K x N) is stored TRANSPOSED as row-major
// (N x K) with stride ldk: per lane 16 contiguous K values of one column.
__device__ __forceinline__ v16bf load_bt_frag(const bf16_t* src, size_t ldk,
                                              int k0, int col0, int lane) {
  const int col = col0 + (lane & 15);
  const int kb  = k0 + ((lane >> 4) << 4);
  return load_contig16(src + (size_t)col * ldk + kb);
}

// ---------------------------------------------------------------------------
// Tensor Data Mover: 2D tile (tile1 rows x tile0 elems, bf16) -> LDS (linear)
// ---------------------------------------------------------------------------
__device__ __forceinline__ void tdm_load_2d(void* lds_ptr, const void* gptr,
                                            unsigned dim0, unsigned dim1,
                                            unsigned tile0, unsigned tile1,
                                            unsigned stride0) {
#if USE_TDM
  const unsigned long long ga = (unsigned long long)(size_t)gptr;
  const unsigned lo = (unsigned)(size_t)lds_ptr;  // flat addr[31:0] == LDS off
  v4u g0;
  g0[0] = 1u;                                   // count=1, user descriptor
  g0[1] = lo;                                   // lds_addr
  g0[2] = (unsigned)ga;                         // global_addr[31:0]
  g0[3] = (unsigned)(ga >> 32) | (2u << 30);    // global_addr[56:32] | type=2
  v8i g1;
  g1[0] = (int)(1u << 16);                      // data_size = 1 -> 2 bytes
  g1[1] = (int)((dim0 & 0xFFFFu) << 16);        // tensor_dim0[15:0]
  g1[2] = (int)((dim0 >> 16) | ((dim1 & 0xFFFFu) << 16));
  g1[3] = (int)((dim1 >> 16) | (tile0 << 16));  // tile_dim0
  g1[4] = (int)(tile1 & 0xFFFFu);               // tile_dim1 (tile_dim2 = 0)
  g1[5] = (int)stride0;                         // tensor_dim0_stride[31:0]
  g1[6] = 0; g1[7] = 0;
  v4i z4 = {0, 0, 0, 0};
  v8i z8 = {0, 0, 0, 0, 0, 0, 0, 0};
  __builtin_amdgcn_tensor_load_to_lds(g0, g1, z4, z4, z8, 0);
  __builtin_amdgcn_s_wait_tensorcnt(0);
#else
  (void)lds_ptr; (void)gptr; (void)dim0; (void)dim1;
  (void)tile0; (void)tile1; (void)stride0;
#endif
}

// ---------------------------------------------------------------------------
// conversions: f32->bf16, f32->bf16 transposed (small), tiled transpose (big)
// ---------------------------------------------------------------------------
__global__ void cvt_bf16_kernel(const float* __restrict__ src,
                                bf16_t* __restrict__ dst, int n) {
  int i = blockIdx.x * blockDim.x + threadIdx.x;
  const int stride = gridDim.x * blockDim.x;
  for (; i < n; i += stride) dst[i] = (bf16_t)src[i];
}

// src (K x N) f32 -> dst (N x K) bf16, coalesced on the write side
__global__ void cvt_t_kernel(const float* __restrict__ src,
                             bf16_t* __restrict__ dst, int K, int N) {
  int i = blockIdx.x * blockDim.x + threadIdx.x;
  const int stride = gridDim.x * blockDim.x;
  const int n = K * N;
  for (; i < n; i += stride) {
    const int nn = i / K, kk = i % K;
    dst[i] = (bf16_t)src[kk * N + nn];
  }
}

// LDS-tiled transpose-convert: src (rows x cols) f32 -> dst (cols x rows) bf16
__global__ void __launch_bounds__(256)
transpose_cvt_kernel(const float* __restrict__ src, bf16_t* __restrict__ dst,
                     int rows, int cols) {
  __shared__ float tile[32][33];
  const int c0 = blockIdx.x * 32, r0 = blockIdx.y * 32;
  const int tx = threadIdx.x, ty = threadIdx.y;  // 32 x 8
  for (int i = ty; i < 32; i += 8)
    tile[i][tx] = src[(size_t)(r0 + i) * cols + (c0 + tx)];
  __syncthreads();
  for (int i = ty; i < 32; i += 8)
    dst[(size_t)(c0 + i) * rows + (r0 + tx)] = (bf16_t)tile[tx][i];
}

// ---------------------------------------------------------------------------
// Generic WMMA GEMM: C = epilogue(A @ Bt^T + bias); A strip staged in LDS via
// TDM; Bt is B transposed (N x K) so B fragments are contiguous 32B loads.
// grid = (M/16, N/(16*WPB)); block = 256. act: 0=none, 1=gelu.
// out = (resid ? resid : 0) + act(acc + bias)
// ---------------------------------------------------------------------------
__global__ void __launch_bounds__(256)
gemm_bf16_kernel(const bf16_t* __restrict__ A, const bf16_t* __restrict__ Bt,
                 const float* __restrict__ bias, const float* __restrict__ resid,
                 bf16_t* __restrict__ Cb, int N, int K, int act) {
  __shared__ __align__(32) bf16_t Ald[16 * 768];  // K <= 768
  const int tid  = threadIdx.x;
  const int lane = tid & 31;
  const int wave = tid >> 5;
  const int m0   = blockIdx.x * 16;
  const int n0   = (blockIdx.y * WPB + wave) * 16;

#if USE_TDM
  if (wave == 0)
    tdm_load_2d(Ald, A + (size_t)m0 * K, K, 16, K, 16, K);
#else
  for (int idx = tid; idx < 16 * K; idx += 256)
    Ald[idx] = A[(size_t)(m0 + idx / K) * K + (idx % K)];
#endif
  __syncthreads();

  v8f acc = zero8();
  for (int c = 0; c < K; c += 32) {
    v16bf af = load_a_frag(Ald, K, 0, c, lane);
    v16bf bf = load_bt_frag(Bt, K, c, n0, lane);
    acc = __builtin_amdgcn_wmma_f32_16x16x32_bf16(false, af, false, bf,
                                                  (short)0, acc, false, false);
  }

  const int col   = n0 + (lane & 15);
  const int rbase = m0 + 8 * (lane >> 4);
  const float bcol = bias ? bias[col] : 0.0f;
#pragma unroll
  for (int r = 0; r < 8; ++r) {
    const int row = rbase + r;
    float v = acc[r] + bcol;
    if (act == 1) v = gelu_exact(v);
    if (resid) v += resid[(size_t)row * N + col];
    Cb[(size_t)row * N + col] = (bf16_t)v;
  }
}

// ---------------------------------------------------------------------------
// Flash attention #1 (Hopfield): 16 query rows / block over 32768 keys.
// Waves 0-1 build S = Q K^T / 16 with WMMA; lanes 0-15 run the online softmax;
// all 8 waves accumulate a 96-wide slice of P @ V (V stored dim-major so the
// B fragments are contiguous). Epilogue fuses the sigmoid(alpha) blend and
// also writes X2^T (dim-major) for the second attention's V operand.
// ---------------------------------------------------------------------------
__global__ void __launch_bounds__(256)
flash_hop_kernel(const bf16_t* __restrict__ Qb,  // 4096 x 256 row-major
                 const bf16_t* __restrict__ Kb,  // 32768 x 256 row-major
                 const bf16_t* __restrict__ Vt,  // 768 x 32768 dim-major
                 const float* __restrict__ X, const float* __restrict__ alpha_p,
                 float* __restrict__ X2f, bf16_t* __restrict__ X2b,
                 bf16_t* __restrict__ X2t /* 768 x 4096 */) {
  __shared__ float  Sld[16][32];
  __shared__ __align__(32) bf16_t Pld[16][32];
  __shared__ float  mrow[16], lrow[16], arow[16];

  const int tid  = threadIdx.x;
  const int lane = tid & 31;
  const int wave = tid >> 5;
  const int q0   = blockIdx.x * 16;

  if (tid < 16) { mrow[tid] = -3.0e38f; lrow[tid] = 0.0f; }

  v16bf qf[8];               // 16x256 Q strip in regs (S-producer waves only)
  if (wave < 2) {
#pragma unroll
    for (int c = 0; c < 8; ++c) qf[c] = load_a_frag(Qb, DINNER, q0, 32 * c, lane);
  }
  v8f acc[NTILES];
#pragma unroll
  for (int t = 0; t < NTILES; ++t) acc[t] = zero8();
  __syncthreads();

  const float scale = 0.0625f;  // 1/sqrt(256)
  for (int j = 0; j < MCTX; j += 32) {
    if (wave < 2) {
      v8f s = zero8();
      const int key0 = j + 16 * wave;
#pragma unroll
      for (int c = 0; c < 8; ++c) {
        v16bf kf = load_bt_frag(Kb, DINNER, 32 * c, key0, lane);
        s = __builtin_amdgcn_wmma_f32_16x16x32_bf16(false, qf[c], false, kf,
                                                    (short)0, s, false, false);
      }
      const int col = 16 * wave + (lane & 15);
      const int rb  = 8 * (lane >> 4);
#pragma unroll
      for (int r = 0; r < 8; ++r) Sld[rb + r][col] = s[r] * scale;
    }
    if (j + 32 < MCTX) {  // global_prefetch_b8 hints for next K/V tiles
      __builtin_prefetch(&Kb[(size_t)(j + 32) * DINNER + tid * 8], 0, 1);
      __builtin_prefetch(&Vt[(size_t)(wave * DVSLICE + lane) * MCTX + j + 32], 0, 1);
    }
    __syncthreads();

    if (tid < 16) {           // online softmax for row `tid`
      const float mold = mrow[tid];
      float mx = mold;
#pragma unroll
      for (int k = 0; k < 32; ++k) mx = fmaxf(mx, Sld[tid][k]);
      const float al = __expf(mold - mx);
      float sum = 0.0f;
#pragma unroll
      for (int k = 0; k < 32; ++k) {
        const float p = __expf(Sld[tid][k] - mx);
        Pld[tid][k] = (bf16_t)p;
        sum += p;
      }
      lrow[tid] = lrow[tid] * al + sum;
      mrow[tid] = mx;
      arow[tid] = al;
    }
    __syncthreads();

    {                         // rescale + P@V for this wave's 96-dim slice
      const int rb = 8 * (lane >> 4);
      float asc[8];
#pragma unroll
      for (int r = 0; r < 8; ++r) asc[r] = arow[rb + r];
      v16bf pf = load_a_frag(&Pld[0][0], 32, 0, 0, lane);
#pragma unroll
      for (int t = 0; t < NTILES; ++t) {
#pragma unroll
        for (int r = 0; r < 8; ++r) acc[t][r] *= asc[r];
        v16bf vf = load_bt_frag(Vt, MCTX, j, wave * DVSLICE + t * 16, lane);
        acc[t] = __builtin_amdgcn_wmma_f32_16x16x32_bf16(false, pf, false, vf,
                                                         (short)0, acc[t],
                                                         false, false);
      }
    }
    __syncthreads();
  }

  const float blend = 1.0f / (1.0f + __expf(-alpha_p[0]));
  const int rb = 8 * (lane >> 4);
  const int cb = wave * DVSLICE + (lane & 15);
#pragma unroll
  for (int t = 0; t < NTILES; ++t) {
    const int col = cb + t * 16;
#pragma unroll
    for (int r = 0; r < 8; ++r) {
      const int row = q0 + rb + r;
      const float o  = acc[t][r] / fmaxf(lrow[rb + r], 1e-20f);
      const float x2 = blend * o + (1.0f - blend) * X[(size_t)row * DMODEL + col];
      X2f[(size_t)row * DMODEL + col] = x2;
      X2b[(size_t)row * DMODEL + col] = (bf16_t)x2;
      X2t[(size_t)col * NROWS + row] = (bf16_t)x2;   // dim-major copy
    }
  }
}

// ---------------------------------------------------------------------------
// Flash attention #2 (molecular): softmax(Qm Km^T / sqrt(768)) @ X2.
// Q strip staged in LDS via TDM. Flash's final /l normalization realizes
// softmax followed by the reference's Dinv re-normalization exactly.
// ---------------------------------------------------------------------------
__global__ void __launch_bounds__(256)
flash_mol_kernel(const bf16_t* __restrict__ Qm,  // 4096 x 768 row-major
                 const bf16_t* __restrict__ Km,  // 4096 x 768 row-major
                 const bf16_t* __restrict__ Vt,  // 768 x 4096 dim-major (X2^T)
                 bf16_t* __restrict__ O2b) {
  __shared__ __align__(32) bf16_t Qld[16 * DMODEL];
  __shared__ float  Sld[16][32];
  __shared__ __align__(32) bf16_t Pld[16][32];
  __shared__ float  mrow[16], lrow[16], arow[16];

  const int tid  = threadIdx.x;
  const int lane = tid & 31;
  const int wave = tid >> 5;
  const int q0   = blockIdx.x * 16;

#if USE_TDM
  if (wave == 0)
    tdm_load_2d(Qld, Qm + (size_t)q0 * DMODEL, DMODEL, 16, DMODEL, 16, DMODEL);
#else
  for (int idx = tid; idx < 16 * DMODEL; idx += 256)
    Qld[idx] = Qm[(size_t)(q0 + idx / DMODEL) * DMODEL + idx % DMODEL];
#endif
  if (tid < 16) { mrow[tid] = -3.0e38f; lrow[tid] = 0.0f; }

  v8f acc[NTILES];
#pragma unroll
  for (int t = 0; t < NTILES; ++t) acc[t] = zero8();
  __syncthreads();

  const float scale = 0.03608439182435161f;  // 1/sqrt(768)
  for (int j = 0; j < NROWS; j += 32) {
    if (wave < 2) {
      v8f s = zero8();
      const int key0 = j + 16 * wave;
      for (int c = 0; c < DMODEL / 32; ++c) {
        v16bf af = load_a_frag(Qld, DMODEL, 0, 32 * c, lane);
        v16bf kf = load_bt_frag(Km, DMODEL, 32 * c, key0, lane);
        s = __builtin_amdgcn_wmma_f32_16x16x32_bf16(false, af, false, kf,
                                                    (short)0, s, false, false);
      }
      const int col = 16 * wave + (lane & 15);
      const int rb  = 8 * (lane >> 4);
#pragma unroll
      for (int r = 0; r < 8; ++r) Sld[rb + r][col] = s[r] * scale;
    }
    if (j + 32 < NROWS) {
      __builtin_prefetch(&Km[(size_t)(j + 32) * DMODEL + tid * 8], 0, 1);
      __builtin_prefetch(&Vt[(size_t)(wave * DVSLICE + lane) * NROWS + j + 32], 0, 1);
    }
    __syncthreads();

    if (tid < 16) {
      const float mold = mrow[tid];
      float mx = mold;
#pragma unroll
      for (int k = 0; k < 32; ++k) mx = fmaxf(mx, Sld[tid][k]);
      const float al = __expf(mold - mx);
      float sum = 0.0f;
#pragma unroll
      for (int k = 0; k < 32; ++k) {
        const float p = __expf(Sld[tid][k] - mx);
        Pld[tid][k] = (bf16_t)p;
        sum += p;
      }
      lrow[tid] = lrow[tid] * al + sum;
      mrow[tid] = mx;
      arow[tid] = al;
    }
    __syncthreads();

    {
      const int rb = 8 * (lane >> 4);
      float asc[8];
#pragma unroll
      for (int r = 0; r < 8; ++r) asc[r] = arow[rb + r];
      v16bf pf = load_a_frag(&Pld[0][0], 32, 0, 0, lane);
#pragma unroll
      for (int t = 0; t < NTILES; ++t) {
#pragma unroll
        for (int r = 0; r < 8; ++r) acc[t][r] *= asc[r];
        v16bf vf = load_bt_frag(Vt, NROWS, j, wave * DVSLICE + t * 16, lane);
        acc[t] = __builtin_amdgcn_wmma_f32_16x16x32_bf16(false, pf, false, vf,
                                                         (short)0, acc[t],
                                                         false, false);
      }
    }
    __syncthreads();
  }

  const int rb = 8 * (lane >> 4);
  const int cb = wave * DVSLICE + (lane & 15);
#pragma unroll
  for (int t = 0; t < NTILES; ++t) {
    const int col = cb + t * 16;
#pragma unroll
    for (int r = 0; r < 8; ++r) {
      const int row = q0 + rb + r;
      O2b[(size_t)row * DMODEL + col] =
          (bf16_t)(acc[t][r] / fmaxf(lrow[rb + r], 1e-20f));
    }
  }
}

// ---------------------------------------------------------------------------
// Final 256->4 layer + distribution heads (tiny; scalar f32).
// ---------------------------------------------------------------------------
__global__ void heads_kernel(const bf16_t* __restrict__ h2,
                             const float* __restrict__ W3,
                             const float* __restrict__ b3,
                             float* __restrict__ out, int nrows) {
  const int i = blockIdx.x * blockDim.x + threadIdx.x;
  if (i >= nrows) return;
  float r0 = 0.f, r1 = 0.f, r2 = 0.f, r3 = 0.f;
  for (int k = 0; k < 256; ++k) {
    const float h = (float)h2[i * 256 + k];
    r0 += h * W3[k * 4 + 0];
    r1 += h * W3[k * 4 + 1];
    r2 += h * W3[k * 4 + 2];
    r3 += h * W3[k * 4 + 3];
  }
  r0 += b3[0]; r1 += b3[1]; r2 += b3[2]; r3 += b3[3];
  out[i]             = r0;
  out[nrows + i]     = softplus_f(r1) + 1e-6f;
  out[2 * nrows + i] = fminf(softplus_f(r2), 28.0f) + 1.01f;
  out[3 * nrows + i] = softplus_f(r3) + 1e-6f;
}

// ---------------------------------------------------------------------------
extern "C" void kernel_launch(void* const* d_in, const int* in_sizes, int n_in,
                              void* d_out, int out_size, void* d_ws, size_t ws_size,
                              hipStream_t stream) {
  const float* X      = (const float*)d_in[0];
  const float* ctx_k  = (const float*)d_in[1];
  const float* ctx_v  = (const float*)d_in[2];
  const float* Wq_hop = (const float*)d_in[3];
  const float* alpha  = (const float*)d_in[4];
  const float* Wq_mol = (const float*)d_in[5];
  const float* Wk_mol = (const float*)d_in[6];
  const float* Wg     = (const float*)d_in[7];
  const float* bg     = (const float*)d_in[8];
  const float* W1     = (const float*)d_in[9];
  const float* b1     = (const float*)d_in[10];
  const float* W2     = (const float*)d_in[11];
  const float* b2     = (const float*)d_in[12];
  const float* W3     = (const float*)d_in[13];
  const float* b3     = (const float*)d_in[14];
  float* out = (float*)d_out;

  char* ws = (char*)d_ws;
  size_t off = 0;
  auto wsalloc = [&](size_t bytes) -> void* {
    void* p = ws + off;
    off += (bytes + 255) & ~size_t(255);
    return p;
  };

  bf16_t* xbf   = (bf16_t*)wsalloc((size_t)NROWS * DMODEL * 2);
  bf16_t* kbf   = (bf16_t*)wsalloc((size_t)MCTX * DINNER * 2);
  bf16_t* vbt   = (bf16_t*)wsalloc((size_t)DMODEL * MCTX * 2);   // ctx_v^T
  bf16_t* wqhT  = (bf16_t*)wsalloc((size_t)DINNER * DMODEL * 2); // (256x768)
  bf16_t* wqmT  = (bf16_t*)wsalloc((size_t)DMODEL * DMODEL * 2);
  bf16_t* wkmT  = (bf16_t*)wsalloc((size_t)DMODEL * DMODEL * 2);
  bf16_t* wgT   = (bf16_t*)wsalloc((size_t)DMODEL * DMODEL * 2);
  bf16_t* w1T   = (bf16_t*)wsalloc((size_t)512 * DMODEL * 2);    // (512x768)
  bf16_t* w2T   = (bf16_t*)wsalloc((size_t)256 * 512 * 2);       // (256x512)
  bf16_t* qbf   = (bf16_t*)wsalloc((size_t)NROWS * DINNER * 2);
  float*  x2f   = (float*) wsalloc((size_t)NROWS * DMODEL * 4);
  bf16_t* x2b   = (bf16_t*)wsalloc((size_t)NROWS * DMODEL * 2);
  bf16_t* x2t   = (bf16_t*)wsalloc((size_t)DMODEL * NROWS * 2);  // X2^T
  bf16_t* qmb   = (bf16_t*)wsalloc((size_t)NROWS * DMODEL * 2);
  bf16_t* kmb   = (bf16_t*)wsalloc((size_t)NROWS * DMODEL * 2);
  bf16_t* o2b   = (bf16_t*)wsalloc((size_t)NROWS * DMODEL * 2);
  bf16_t* hb    = (bf16_t*)wsalloc((size_t)NROWS * DMODEL * 2);
  bf16_t* h1b   = (bf16_t*)wsalloc((size_t)NROWS * 512 * 2);
  bf16_t* h2b   = (bf16_t*)wsalloc((size_t)NROWS * 256 * 2);
  (void)ws_size; (void)in_sizes; (void)n_in; (void)out_size;

  auto cvt = [&](const float* s, bf16_t* d, int n) {
    int blocks = (n + 255) / 256; if (blocks > 2048) blocks = 2048;
    cvt_bf16_kernel<<<blocks, 256, 0, stream>>>(s, d, n);
  };
  auto cvtT = [&](const float* s, bf16_t* d, int K, int N) {
    int n = K * N;
    int blocks = (n + 255) / 256; if (blocks > 2048) blocks = 2048;
    cvt_t_kernel<<<blocks, 256, 0, stream>>>(s, d, K, N);
  };

  cvt(X, xbf, NROWS * DMODEL);
  cvt(ctx_k, kbf, MCTX * DINNER);
  transpose_cvt_kernel<<<dim3(DMODEL / 32, MCTX / 32), dim3(32, 8), 0, stream>>>(
      ctx_v, vbt, MCTX, DMODEL);
  cvtT(Wq_hop, wqhT, DMODEL, DINNER);
  cvtT(Wq_mol, wqmT, DMODEL, DMODEL);
  cvtT(Wk_mol, wkmT, DMODEL, DMODEL);
  cvtT(Wg, wgT, DMODEL, DMODEL);
  cvtT(W1, w1T, DMODEL, 512);
  cvtT(W2, w2T, 512, 256);

  auto gemm = [&](const bf16_t* A, const bf16_t* Bt, const float* bias,
                  const float* resid, bf16_t* Cb, int N, int K, int act) {
    dim3 grid(NROWS / 16, N / (16 * WPB));
    gemm_bf16_kernel<<<grid, 256, 0, stream>>>(A, Bt, bias, resid, Cb, N, K, act);
  };

  // Q = X @ Wq_hop
  gemm(xbf, wqhT, nullptr, nullptr, qbf, DINNER, DMODEL, 0);
  // enriched = softmax(Q K^T/16) @ V ; X2 = blend*enriched + (1-blend)*X
  flash_hop_kernel<<<NROWS / 16, 256, 0, stream>>>(qbf, kbf, vbt, X, alpha,
                                                   x2f, x2b, x2t);
  // Qm, Km
  gemm(x2b, wqmT, nullptr, nullptr, qmb, DMODEL, DMODEL, 0);
  gemm(x2b, wkmT, nullptr, nullptr, kmb, DMODEL, DMODEL, 0);
  // O2 = normalized-softmax(Qm Km^T / sqrt(768)) @ X2
  flash_mol_kernel<<<NROWS / 16, 256, 0, stream>>>(qmb, kmb, x2t, o2b);
  // h = X2 + gelu(O2 @ Wg + bg)
  gemm(o2b, wgT, bg, x2f, hb, DMODEL, DMODEL, 1);
  // MLP
  gemm(hb, w1T, b1, nullptr, h1b, 512, DMODEL, 1);
  gemm(h1b, w2T, b2, nullptr, h2b, 256, 512, 1);
  // heads
  heads_kernel<<<NROWS / 256, 256, 0, stream>>>(h2b, W3, b3, out, NROWS);
}
